// SPE_Module_84155589197863
// MI455X (gfx1250) — matrix operand order repaired and checked by reading, not compiled, not run on hardware
//
#include <hip/hip_runtime.h>

// ---------- types ----------
typedef __attribute__((ext_vector_type(16))) __bf16        v16bf;
typedef __attribute__((ext_vector_type(8)))  float         v8f;
typedef __attribute__((ext_vector_type(4)))  unsigned int  u32x4;
typedef __attribute__((ext_vector_type(8)))  unsigned int  u32x8;
typedef __attribute__((ext_vector_type(4)))  float         f32x4;

#define L_DIM  128
#define N_ROWS 65536
#define B_DIM  8
#define SLICES 128
#define NC     (N_ROWS / SLICES)   // 512 rows of q per block
#define CHUNK  32                  // K per WMMA step
#define NCH    (NC / CHUNK)        // 16 chunks

union Tile16 { u32x4 q[2]; unsigned short s[16]; v16bf v; };
static_assert(sizeof(Tile16) == 32, "tile size");

__device__ __forceinline__ unsigned short f2bf(float f) {
  union { float f; unsigned int u; } c; c.f = f;
  unsigned int u = c.u + 0x7FFFu + ((c.u >> 16) & 1u);  // round-to-nearest-even
  return (unsigned short)(u >> 16);
}

// ---------- kernel 0: zero the energy accumulator ----------
__global__ __launch_bounds__(256) void zero_kernel(float* __restrict__ p, int n) {
  int i = blockIdx.x * 256 + threadIdx.x;
  if (i < n) p[i] = 0.0f;
}

// ---------- kernel 1: partial Gram  energy[b] += Q_sliceT * Q_slice ----------
// Double-buffered LDS ping-pong: issue next chunk's global loads before the
// WMMAs on the current buffer; convert+store after; one barrier per chunk.
__global__ __launch_bounds__(256) void gram_kernel(const float* __restrict__ x,
                                                   float* __restrict__ energy) {
  __shared__ __align__(16) unsigned short ldsQ[2][L_DIM][CHUNK];  // 2 x 8 KB

  const int b     = blockIdx.y;
  const int slice = blockIdx.x;
  const int t     = threadIdx.x;
  const int wid   = t >> 5;        // 8 waves
  const int lane  = t & 31;
  const int hl    = lane & 15;
  const bool hi   = (lane >= 16);

  const int rr = t >> 5;           // staging row within chunk group
  const int c0 = (t & 31) * 4;     // staging col

  v8f acc[8] = {};

  const size_t batchBase = (size_t)b * N_ROWS * L_DIM;
  const int n0 = slice * NC;

  // prologue: stage chunk 0 into buffer 0
  {
#pragma unroll
    for (int p = 0; p < 4; ++p) {
      const int r = p * 8 + rr;
      f32x4 v = *(const f32x4*)&x[batchBase + (size_t)(n0 + r) * L_DIM + c0];
      ldsQ[0][c0 + 0][r] = f2bf(v.x);
      ldsQ[0][c0 + 1][r] = f2bf(v.y);
      ldsQ[0][c0 + 2][r] = f2bf(v.z);
      ldsQ[0][c0 + 3][r] = f2bf(v.w);
    }
  }
  __syncthreads();

  for (int ch = 0; ch < NCH; ++ch) {
    const int cur  = ch & 1;
    const bool more = (ch + 1 < NCH);

    // issue next chunk's global loads early (latency hidden under WMMAs)
    f32x4 stage[4];
    if (more) {
      const int nb = n0 + (ch + 1) * CHUNK;
      if (ch + 2 < NCH)
        __builtin_prefetch(&x[batchBase + (size_t)(n0 + (ch + 2) * CHUNK) * L_DIM], 0, 0);
#pragma unroll
      for (int p = 0; p < 4; ++p) {
        const int r = p * 8 + rr;
        stage[p] = *(const f32x4*)&x[batchBase + (size_t)(nb + r) * L_DIM + c0];
      }
    }

    // A = Q^T tile 16x32 (M=l, K=n). ISA 16-bit A layout:
    // lanes 0-15: K = 0..7,16..23 ; lanes 16-31: K = 8..15,24..31
    const int kbA  = hi ? 8 : 0;
    const int lrow = wid * 16 + hl;
    Tile16 A;
    A.q[0] = *(const u32x4*)&ldsQ[cur][lrow][kbA];
    A.q[1] = *(const u32x4*)&ldsQ[cur][lrow][kbA + 16];

    // preload all 8 B tiles into distinct registers -> single DScnt wait,
    // then 8 back-to-back WMMAs
    const int kbB = hi ? 8 : 0;
    Tile16 Bt[8];
#pragma unroll
    for (int m = 0; m < 8; ++m) {
      const int nc = m * 16 + hl;
      Bt[m].q[0] = *(const u32x4*)&ldsQ[cur][nc][kbB];
      Bt[m].q[1] = *(const u32x4*)&ldsQ[cur][nc][kbB + 16];
    }
#pragma unroll
    for (int m = 0; m < 8; ++m)
      acc[m] = __builtin_amdgcn_wmma_f32_16x16x32_bf16(
          false, A.v, false, Bt[m].v, (short)0, acc[m], false, false);

    // convert + store next chunk into the other buffer (no one reads it yet)
    if (more) {
      const int nxt = cur ^ 1;
#pragma unroll
      for (int p = 0; p < 4; ++p) {
        const int r = p * 8 + rr;
        ldsQ[nxt][c0 + 0][r] = f2bf(stage[p].x);
        ldsQ[nxt][c0 + 1][r] = f2bf(stage[p].y);
        ldsQ[nxt][c0 + 2][r] = f2bf(stage[p].z);
        ldsQ[nxt][c0 + 3][r] = f2bf(stage[p].w);
      }
    }
    __syncthreads();
  }

  // merge partials: D layout VGPR r -> M = r + (hi?8:0), N = hl
#pragma unroll
  for (int m = 0; m < 8; ++m) {
#pragma unroll
    for (int r = 0; r < 8; ++r) {
      const int row = wid * 16 + r + (hi ? 8 : 0);
      const int col = m * 16 + hl;
      unsafeAtomicAdd(&energy[((size_t)b * L_DIM + row) * L_DIM + col], acc[m][r]);
    }
  }
}

// ---------- kernel 2: row softmax, store attn TRANSPOSED as bf16 ----------
__global__ __launch_bounds__(128) void softmax_kernel(const float* __restrict__ energy,
                                                      unsigned short* __restrict__ attnT) {
  __shared__ float red[L_DIM];
  const int l = blockIdx.x;     // row of energy
  const int b = blockIdx.y;
  const int t = threadIdx.x;    // column

  const float e = energy[((size_t)b * L_DIM + l) * L_DIM + t];
  red[t] = e;
  __syncthreads();
#pragma unroll
  for (int s = 64; s > 0; s >>= 1) {
    if (t < s) red[t] = fmaxf(red[t], red[t + s]);
    __syncthreads();
  }
  const float mx = red[0];
  __syncthreads();
  const float ex = __expf(e - mx);
  red[t] = ex;
  __syncthreads();
#pragma unroll
  for (int s = 64; s > 0; s >>= 1) {
    if (t < s) red[t] += red[t + s];
    __syncthreads();
  }
  const float inv = __frcp_rn(red[0]);
  // transposed store: attnT[b][m=t][l]
  attnT[(size_t)b * L_DIM * L_DIM + (size_t)t * L_DIM + l] = f2bf(ex * inv);
}

// ---------- kernel 3: out = Q * attn ----------
// attn^T (128x128 bf16 = 32 KB) is pulled into LDS by the Tensor Data Mover
// (one DMA per block, issued by wave 0), then 8 waves run bf16 WMMAs.
__global__ __launch_bounds__(256) void av_kernel(const float* __restrict__ x,
                                                 const unsigned short* __restrict__ attnT,
                                                 float* __restrict__ out) {
  __shared__ __align__(16) unsigned short lAttn[L_DIM * L_DIM];  // 32 KB, [m][l]

  const int b    = blockIdx.y;
  const int t    = threadIdx.x;
  const int wid  = t >> 5;
  const int lane = t & 31;
  const int hl   = lane & 15;
  const bool hi  = (lane >= 16);

  // --- TDM: tensor_load_to_lds of the whole 128x128 bf16 tile ---
  if (wid == 0) {
    const unsigned long long gaddr =
        (unsigned long long)(const void*)(attnT + (size_t)b * L_DIM * L_DIM);
    const unsigned ldsBase = (unsigned)(size_t)(void*)&lAttn[0];

    // D# group 0: count=1 | lds_addr | global_addr[56:0] | type=2
    u32x4 g0;
    g0.x = 1u;                                   // count=1, user mode
    g0.y = ldsBase;                              // LDS byte address
    g0.z = (unsigned)(gaddr & 0xFFFFFFFFu);      // global_addr[31:0]
    g0.w = (unsigned)((gaddr >> 32) & 0x01FFFFFFu) | (2u << 30);  // [56:32] | type=2

    // D# group 1: data_size=2B, tensor 128x128, tile 128x128, stride0=128
    u32x8 g1;
    g1.s0 = (1u << 16);          // workgroup_mask=0, data_size=1 (2 bytes)
    g1.s1 = (128u << 16);        // tensor_dim0[15:0]=128 (atomic_barrier_addr=0)
    g1.s2 = (128u << 16);        // tensor_dim0[31:16]=0, tensor_dim1[15:0]=128
    g1.s3 = (128u << 16);        // tensor_dim1[31:16]=0, tile_dim0=128
    g1.s4 = 128u;                // tile_dim1=128, tile_dim2=0
    g1.s5 = 128u;                // tensor_dim0_stride[31:0]=128
    g1.s6 = 0u;                  // stride0 hi / tensor_dim1_stride lo
    g1.s7 = 0u;                  // tensor_dim1_stride hi
    u32x4 gz = {0u, 0u, 0u, 0u};  // groups 2/3: tile_dim3/4 = 0 (unused)

    asm volatile("tensor_load_to_lds %0, %1, %2, %3"
                 :
                 : "s"(g0), "s"(g1), "s"(gz), "s"(gz)
                 : "memory");
    __builtin_amdgcn_s_wait_tensorcnt(0);
  }
  __syncthreads();

  const int nbase = blockIdx.x * 128 + wid * 16;
  const int n     = nbase + hl;
  const float* xrow = &x[((size_t)b * N_ROWS + n) * (size_t)L_DIM];

  v8f acc[8] = {};

#pragma unroll
  for (int kc = 0; kc < 4; ++kc) {          // l-chunks of 32
    // A = q rows tile 16x32, fp32 global -> bf16 registers
    const int kbA = hi ? 8 : 0;
    const float* ap = xrow + kc * 32 + kbA;
    f32x4 a0 = *(const f32x4*)(ap + 0);
    f32x4 a1 = *(const f32x4*)(ap + 4);
    f32x4 a2 = *(const f32x4*)(ap + 16);
    f32x4 a3 = *(const f32x4*)(ap + 20);
    const float af[16] = {a0.x, a0.y, a0.z, a0.w, a1.x, a1.y, a1.z, a1.w,
                          a2.x, a2.y, a2.z, a2.w, a3.x, a3.y, a3.z, a3.w};
    Tile16 A;
#pragma unroll
    for (int i = 0; i < 16; ++i) A.s[i] = f2bf(af[i]);

    // preload 8 B tiles, then back-to-back WMMAs
    const int kbB = hi ? 8 : 0;
    Tile16 Bt[8];
#pragma unroll
    for (int m = 0; m < 8; ++m) {
      const unsigned short* bp = &lAttn[(m * 16 + hl) * L_DIM + kc * 32 + kbB];
      Bt[m].q[0] = ((const u32x4*)bp)[0];
      Bt[m].q[1] = ((const u32x4*)bp)[2];
    }
#pragma unroll
    for (int m = 0; m < 8; ++m)
      acc[m] = __builtin_amdgcn_wmma_f32_16x16x32_bf16(
          false, A.v, false, Bt[m].v, (short)0, acc[m], false, false);
  }

#pragma unroll
  for (int m = 0; m < 8; ++m) {
#pragma unroll
    for (int r = 0; r < 8; ++r) {
      const int row = nbase + r + (hi ? 8 : 0);
      out[((size_t)b * N_ROWS + row) * L_DIM + m * 16 + hl] = acc[m][r];
    }
  }
}

// ---------- launcher ----------
extern "C" void kernel_launch(void* const* d_in, const int* in_sizes, int n_in,
                              void* d_out, int out_size, void* d_ws, size_t ws_size,
                              hipStream_t stream) {
  (void)in_sizes; (void)n_in; (void)out_size; (void)ws_size;
  const float* x = (const float*)d_in[0];
  float* out = (float*)d_out;

  float* energy = (float*)d_ws;                                       // 512 KB
  unsigned short* attnT =
      (unsigned short*)((char*)d_ws + (size_t)B_DIM * L_DIM * L_DIM * sizeof(float));  // 256 KB

  const int nEnergy = B_DIM * L_DIM * L_DIM;  // 131072
  zero_kernel<<<(nEnergy + 255) / 256, 256, 0, stream>>>(energy, nEnergy);
  gram_kernel<<<dim3(SLICES, B_DIM), 256, 0, stream>>>(x, energy);
  softmax_kernel<<<dim3(L_DIM, B_DIM), 128, 0, stream>>>(energy, attnT);
  av_kernel<<<dim3(N_ROWS / 128, B_DIM), 256, 0, stream>>>(x, attnT, out);
}
